// MultiHeadAttention_53901839565716
// MI455X (gfx1250) — matrix operand order repaired and checked by reading, not compiled, not run on hardware
//
#include <hip/hip_runtime.h>
#include <hip/hip_bf16.h>

typedef __attribute__((ext_vector_type(16))) _Float16 v16h;
typedef __attribute__((ext_vector_type(8)))  float    v8f;

constexpr int B_ = 4, L_ = 2048, D_ = 512, H_ = 8, DH_ = 64;
constexpr int G_ = B_ * H_;        // 32 attention groups (flat reshape, no transpose)
constexpr int M_ = B_ * L_;        // 8192 projection rows
constexpr float SCALE_ = 8.0f;     // DH // H, multiplicative (faithful to source)
constexpr float LN_EPS_ = 1e-5f;

__device__ __forceinline__ int lane_id() { return threadIdx.x & 31; }

// ---------- WMMA 16x16x32 f16 fragment helpers (wave32, gfx1250 layouts) ----------
// A-matrix 16(M)x32(K): lane l -> row m=l&15, half h=l>>4.
// element i: k = i + 8h for i<8 ; k = i + 8 + 8h for i>=8  (ISA 7.12.2)
__device__ __forceinline__ v16h load_a(const float* src, int ld) {
  int l = lane_id();
  const float* p = src + (l & 15) * ld + ((l >> 4) << 3);
  v16h a;
#pragma unroll
  for (int i = 0; i < 8; ++i) { a[i] = (_Float16)p[i]; a[i + 8] = (_Float16)p[i + 16]; }
  return a;
}
__device__ __forceinline__ v16h load_a(const _Float16* src, int ld) {
  int l = lane_id();
  const _Float16* p = src + (l & 15) * ld + ((l >> 4) << 3);
  v16h a;
#pragma unroll
  for (int i = 0; i < 8; ++i) { a[i] = p[i]; a[i + 8] = p[i + 16]; }
  return a;
}
// B-matrix 32(K)x16(N) from a TRANSPOSED row-major source (srcT[n][k], stride ld):
// lane l -> col n=l&15, half h=l>>4; element i: k = i + 8h for i<8 ; k = i + 8 + 8h for i>=8
__device__ __forceinline__ v16h load_bT(const float* srcT, int ld) {
  int l = lane_id();
  const float* p = srcT + (l & 15) * ld + ((l >> 4) << 3);
  v16h b;
#pragma unroll
  for (int i = 0; i < 8; ++i) { b[i] = (_Float16)p[i]; b[i + 8] = (_Float16)p[i + 16]; }
  return b;
}
__device__ __forceinline__ v16h load_bT(const _Float16* srcT, int ld) {
  int l = lane_id();
  const _Float16* p = srcT + (l & 15) * ld + ((l >> 4) << 3);
  v16h b;
#pragma unroll
  for (int i = 0; i < 8; ++i) { b[i] = p[i]; b[i + 8] = p[i + 16]; }
  return b;
}
__device__ __forceinline__ v8f wmma32(v16h a, v16h b, v8f c) {
  return __builtin_amdgcn_wmma_f32_16x16x32_f16(false, a, false, b, (short)0, c,
                                                false, false);
}

// ---------- Y = X @ W^T + bias ; one 16x64 panel per wave (A reused 4x) ----------
template <typename TA, typename TO>
__global__ void gemm_wT_bias(const TA* __restrict__ X, const float* __restrict__ W,
                             const float* __restrict__ bias, TO* __restrict__ Y,
                             int Mm, int Nn, int Kk) {
  int wid = blockIdx.x * (blockDim.x >> 5) + (threadIdx.x >> 5);
  int Nt = Nn >> 6;                                  // 64-wide N panels
  int mt = wid / Nt, np = wid % Nt;
  const TA*    xrow = X + (size_t)mt * 16 * Kk;
  const float* w0 = W + (size_t)np * 64 * Kk;        // rows of W are columns of W^T
  const float* w1 = w0 + (size_t)16 * Kk;
  const float* w2 = w0 + (size_t)32 * Kk;
  const float* w3 = w0 + (size_t)48 * Kk;
  v8f acc0 = {}, acc1 = {}, acc2 = {}, acc3 = {};
  for (int k0 = 0; k0 < Kk; k0 += 32) {
    v16h a = load_a(xrow + k0, Kk);
    acc0 = wmma32(a, load_bT(w0 + k0, Kk), acc0);
    acc1 = wmma32(a, load_bT(w1 + k0, Kk), acc1);
    acc2 = wmma32(a, load_bT(w2 + k0, Kk), acc2);
    acc3 = wmma32(a, load_bT(w3 + k0, Kk), acc3);
  }
  int l = lane_id(), n = l & 15, h = l >> 4;
  int col = np * 64 + n;
  float b0 = bias[col], b1 = bias[col + 16], b2 = bias[col + 32], b3 = bias[col + 48];
  TO* yp = Y + ((size_t)mt * 16 + h * 8) * Nn + col;
#pragma unroll
  for (int r = 0; r < 8; ++r) {
    yp[(size_t)r * Nn]      = (TO)(acc0[r] + b0);
    yp[(size_t)r * Nn + 16] = (TO)(acc1[r] + b1);
    yp[(size_t)r * Nn + 32] = (TO)(acc2[r] + b2);
    yp[(size_t)r * Nn + 48] = (TO)(acc3[r] + b3);
  }
}

// ---------- vT[g][n][k] = v[g][k][n] via LDS tile transpose (64x64 f16 tiles) ----------
__global__ void transpose_v(const _Float16* __restrict__ v, _Float16* __restrict__ vT) {
  __shared__ _Float16 tile[64][66];
  int g  = blockIdx.x >> 5;            // 0..31
  int k0 = (blockIdx.x & 31) * 64;     // 64-row strip of the [2048 x 64] panel
  int t = threadIdx.x;
  int r = t >> 2, c0 = (t & 3) * 16;
  const _Float16* src = v + ((size_t)g * L_ + k0) * DH_;
#pragma unroll
  for (int i = 0; i < 16; ++i) tile[r][c0 + i] = src[(size_t)r * DH_ + c0 + i];
  __syncthreads();
  _Float16* dst = vT + ((size_t)g * DH_ + r) * L_ + k0 + c0;   // row n=r of vT
#pragma unroll
  for (int i = 0; i < 16; ++i) dst[i] = tile[c0 + i][r];
}

// ---------- scores (SCALE * Q K^T) + row softmax, fused per 16-row block ----------
__global__ void scores_softmax(const _Float16* __restrict__ q,
                               const _Float16* __restrict__ k,
                               float* __restrict__ attn) {
  int g  = blockIdx.x >> 7;     // 0..31
  int mt = blockIdx.x & 127;    // 0..127
  const _Float16* qbase = q + ((size_t)g * L_ + mt * 16) * DH_;
  const _Float16* kbase = k + (size_t)g * L_ * DH_;
  float* abase = attn + ((size_t)g * L_ + mt * 16) * (size_t)L_;

  // Phase 1: 8 waves x 16 column tiles each; K=64 -> 2 WMMAs per tile
  int w = threadIdx.x >> 5;
  v16h a0 = load_a(qbase, DH_);
  v16h a1 = load_a(qbase + 32, DH_);
  int l = lane_id(), n = l & 15, h = l >> 4;
  for (int jt = w; jt < L_ / 16; jt += 8) {
    const _Float16* kb = kbase + (size_t)jt * 16 * DH_;
    v8f acc = {};
    acc = wmma32(a0, load_bT(kb, DH_), acc);
    acc = wmma32(a1, load_bT(kb + 32, DH_), acc);
    float* sp = abase + (size_t)(h * 8) * L_ + jt * 16 + n;
#pragma unroll
    for (int r = 0; r < 8; ++r) sp[(size_t)r * L_] = acc[r] * SCALE_;
  }
  __threadfence_block();
  __syncthreads();

  // Phase 2: softmax over each of the 16 rows (2048 cols, L2-hot); 16 threads/row
  __shared__ float red[16][17];
  int t = threadIdx.x, r = t >> 4, s = t & 15;
  float* rp = abase + (size_t)r * L_ + s * 128;
  float mx = -3.402823466e38f;
  for (int i = 0; i < 128; ++i) mx = fmaxf(mx, rp[i]);
  red[r][s] = mx;
  __syncthreads();
  float rowmax = red[r][0];
#pragma unroll
  for (int i = 1; i < 16; ++i) rowmax = fmaxf(rowmax, red[r][i]);
  __syncthreads();
  float sum = 0.f;
  for (int i = 0; i < 128; ++i) sum += __expf(rp[i] - rowmax);
  red[r][s] = sum;
  __syncthreads();
  float rowsum = 0.f;
#pragma unroll
  for (int i = 0; i < 16; ++i) rowsum += red[r][i];
  float inv = 1.0f / rowsum;
  for (int i = 0; i < 128; ++i) rp[i] = __expf(rp[i] - rowmax) * inv;
}

// ---------- ctx = attn @ v ; one 16x64 tile per wave, K=2048, vT gives contiguous B ----------
__global__ void attn_v(const float* __restrict__ attn, const _Float16* __restrict__ vT,
                       _Float16* __restrict__ ctx) {
  int wid = blockIdx.x * 8 + (threadIdx.x >> 5);   // 0..4095
  int g = wid >> 7, mt = wid & 127;
  const float*    abase = attn + ((size_t)g * L_ + mt * 16) * (size_t)L_;
  const _Float16* vt0 = vT + (size_t)g * DH_ * L_;   // vT[g][n][k], row n contiguous in k
  const _Float16* vt1 = vt0 + (size_t)16 * L_;
  const _Float16* vt2 = vt0 + (size_t)32 * L_;
  const _Float16* vt3 = vt0 + (size_t)48 * L_;
  v8f acc0 = {}, acc1 = {}, acc2 = {}, acc3 = {};
  for (int k0 = 0; k0 < L_; k0 += 32) {
    __builtin_prefetch(abase + k0 + 64, 0, 0);       // global_prefetch_b8
    v16h a = load_a(abase + k0, L_);                 // f32 probs -> f16 on the fly
    acc0 = wmma32(a, load_bT(vt0 + k0, L_), acc0);
    acc1 = wmma32(a, load_bT(vt1 + k0, L_), acc1);
    acc2 = wmma32(a, load_bT(vt2 + k0, L_), acc2);
    acc3 = wmma32(a, load_bT(vt3 + k0, L_), acc3);
  }
  int l = lane_id(), n = l & 15, h = l >> 4;
  _Float16* cp = ctx + ((size_t)g * L_ + mt * 16 + h * 8) * DH_ + n;
#pragma unroll
  for (int r = 0; r < 8; ++r) {
    cp[(size_t)r * DH_ +  0] = (_Float16)acc0[r];
    cp[(size_t)r * DH_ + 16] = (_Float16)acc1[r];
    cp[(size_t)r * DH_ + 32] = (_Float16)acc2[r];
    cp[(size_t)r * DH_ + 48] = (_Float16)acc3[r];
  }
}

// ---------- out = LayerNorm(residual + proj) ----------
__global__ void residual_ln(const float* __restrict__ res, const float* __restrict__ proj,
                            const float* __restrict__ gamma, const float* __restrict__ beta,
                            float* __restrict__ out) {
  int row = blockIdx.x, t = threadIdx.x;
  const float* rp = res + (size_t)row * D_;
  const float* pp = proj + (size_t)row * D_;
  float x0 = rp[t] + pp[t];
  float x1 = rp[t + 256] + pp[t + 256];
  __shared__ float s1[256], s2[256];
  s1[t] = x0 + x1;
  s2[t] = x0 * x0 + x1 * x1;
  __syncthreads();
  for (int off = 128; off > 0; off >>= 1) {
    if (t < off) { s1[t] += s1[t + off]; s2[t] += s2[t + off]; }
    __syncthreads();
  }
  float mean = s1[0] * (1.0f / D_);
  float var  = s2[0] * (1.0f / D_) - mean * mean;
  float inv  = rsqrtf(var + LN_EPS_);
  float* op = out + (size_t)row * D_;
  op[t]       = (x0 - mean) * inv * gamma[t]       + beta[t];
  op[t + 256] = (x1 - mean) * inv * gamma[t + 256] + beta[t + 256];
}

extern "C" void kernel_launch(void* const* d_in, const int* in_sizes, int n_in,
                              void* d_out, int out_size, void* d_ws, size_t ws_size,
                              hipStream_t stream) {
  (void)in_sizes; (void)n_in; (void)out_size; (void)ws_size;
  const float* query = (const float*)d_in[0];
  const float* key_  = (const float*)d_in[1];
  const float* value = (const float*)d_in[2];
  const float* Wq = (const float*)d_in[3];
  const float* bq = (const float*)d_in[4];
  const float* Wk = (const float*)d_in[5];
  const float* bk = (const float*)d_in[6];
  const float* Wv = (const float*)d_in[7];
  const float* bv = (const float*)d_in[8];
  const float* Wo = (const float*)d_in[9];
  const float* bo = (const float*)d_in[10];
  const float* gamma = (const float*)d_in[11];
  const float* beta  = (const float*)d_in[12];

  float* out  = (float*)d_out;
  float* attn = out + (size_t)M_ * D_;             // tuple order: (out, attn)

  // workspace: q,k,v,ctx f16 + proj f32 + vT f16  (~58 MB total)
  _Float16* qf = (_Float16*)d_ws;
  _Float16* kf = qf + (size_t)M_ * D_;
  _Float16* vf = kf + (size_t)M_ * D_;
  _Float16* cf = vf + (size_t)M_ * D_;
  float*  proj = (float*)(cf + (size_t)M_ * D_);
  _Float16* vT = (_Float16*)(proj + (size_t)M_ * D_);

  const int gemmBlocks = (M_ / 16) * (D_ / 64) / 8;   // 512 blocks x 8 waves
  gemm_wT_bias<float, _Float16><<<gemmBlocks, 256, 0, stream>>>(query, Wq, bq, qf, M_, D_, D_);
  gemm_wT_bias<float, _Float16><<<gemmBlocks, 256, 0, stream>>>(key_,  Wk, bk, kf, M_, D_, D_);
  gemm_wT_bias<float, _Float16><<<gemmBlocks, 256, 0, stream>>>(value, Wv, bv, vf, M_, D_, D_);
  transpose_v<<<G_ * (L_ / 64), 256, 0, stream>>>(vf, vT);
  scores_softmax<<<G_ * (L_ / 16), 256, 0, stream>>>(qf, kf, attn);
  attn_v<<<G_ * (L_ / 16) / 8, 256, 0, stream>>>(attn, vT, cf);
  gemm_wT_bias<_Float16, float><<<gemmBlocks, 256, 0, stream>>>(cf, Wo, bo, proj, M_, D_, D_);
  residual_ln<<<M_, 256, 0, stream>>>(query, proj, gamma, beta, out);
}
